// S5Layer_33354716021197
// MI455X (gfx1250) — matrix operand order repaired and checked by reading, not compiled, not run on hardware
//
#include <hip/hip_runtime.h>
#include <hip/hip_bf16.h>

// ---------------------------------------------------------------------------
// S5 layer on MI455X (gfx1250, wave32, WMMA 16x16x32 bf16)
// BATCH=64, SEQ=2048, D_MODEL=512, D_STATE=64
// ---------------------------------------------------------------------------

#define BATCH   64
#define SEQ     2048
#define DMODEL  512
#define DSTATE  64
#define ROWS    (BATCH * SEQ)          // 131072 flattened tokens
#define RTILES  (ROWS / 16)            // 8192 16-row WMMA tiles
#define CHUNKS  32
#define CLEN    (SEQ / CHUNKS)         // 64
#define DT_MIN  0.001f
#define DT_MAX  0.1f
#define LN_EPS  1e-5f

typedef __attribute__((ext_vector_type(8)))  float  v8f;
typedef __attribute__((ext_vector_type(16))) __bf16 v16bf;

// ---------------------------------------------------------------------------
// Kernel 0: convert B_mat [64x512] and C_mat [512x64] weights f32 -> bf16
// ---------------------------------------------------------------------------
__global__ void s5_cvt_weights(const float* __restrict__ Bm,
                               const float* __restrict__ Cm,
                               __bf16* __restrict__ Bb,
                               __bf16* __restrict__ Cb) {
    int i = blockIdx.x * blockDim.x + threadIdx.x;
    if (i < DSTATE * DMODEL) {
        Bb[i] = (__bf16)Bm[i];
        Cb[i] = (__bf16)Cm[i];
    }
}

// ---------------------------------------------------------------------------
// Kernel 1: one wave per 16-row tile.
//   acc[16x64] = x_tile[16x512] (bf16) x B^T (bf16)  via v_wmma_f32_16x16x32_bf16
//   dt[row]    = sigmoid(x . dt_w + dt_b) * (DT_MAX-DT_MIN) + DT_MIN   (f32)
//   dA  = exp(dt * (-exp(A_log)))  ;  dBu = dt * acc
// 16-bit A 16x32 / B 32x16 fragments use contiguous k = 16*half + i (same in A and B); f32 C/D.
// ---------------------------------------------------------------------------
__global__ __launch_bounds__(32) void s5_gemm1_dt(
        const float* __restrict__ x,
        const float* __restrict__ A_log,
        const __bf16* __restrict__ Bb,
        const float* __restrict__ dt_w,
        const float* __restrict__ dt_b,
        float* __restrict__ dA,
        float* __restrict__ dBu) {
    const int lane = threadIdx.x;
    const int mrow = lane & 15;
    const int half = lane >> 4;
    const int row0 = blockIdx.x * 16;

    v8f acc[4] = {};           // 4 column tiles of 16 -> N=64
    float dtacc = 0.f;

    const float* xrow = x + (size_t)(row0 + mrow) * DMODEL;

    for (int k = 0; k < DMODEL / 32; ++k) {
        const int kbase = k * 32;
        // A fragment: a[i] = x[M = lane&15][kbase + half*16 + i]
        v16bf a;
#pragma unroll
        for (int i = 0; i < 16; ++i) {
            const int col = kbase + (half << 4) + i;
            const float xv = xrow[col];
            a[i] = (__bf16)xv;
            dtacc += xv * dt_w[col];   // f32 dt dot alongside (halves disjoint in K)
        }
#pragma unroll
        for (int nt = 0; nt < 4; ++nt) {
            // B fragment: b[i] = B_mat[n][kbase + half*16 + i]  (B^T in WMMA terms)
            const int n = nt * 16 + mrow;
            const __bf16* bp = Bb + (size_t)n * DMODEL + kbase + (half << 4);
            v16bf b;
#pragma unroll
            for (int i = 0; i < 16; ++i) b[i] = bp[i];
            acc[nt] = __builtin_amdgcn_wmma_f32_16x16x32_bf16(
                false, a, false, b, (short)0, acc[nt], false, false);
        }
    }

    // combine K-halves: lane L and L+16 each hold half of row (L&15)'s dot
    dtacc += __shfl_xor(dtacc, 16);
    const float z  = dtacc + dt_b[0];
    float dt = 1.f / (1.f + __expf(-z));
    dt = dt * (DT_MAX - DT_MIN) + DT_MIN;          // dt for row (lane&15)

#pragma unroll
    for (int nt = 0; nt < 4; ++nt) {
        const int n = nt * 16 + mrow;
        const float Adiag = -__expf(A_log[n]);
#pragma unroll
        for (int r = 0; r < 8; ++r) {
            const int M = r + (half << 3);
            const float dtm = __shfl(dt, M);       // lane M holds row M's dt
            const size_t g = (size_t)(row0 + M) * DSTATE + n;
            dA[g]  = __expf(dtm * Adiag);
            dBu[g] = dtm * acc[nt][r];
        }
    }
}

// ---------------------------------------------------------------------------
// Kernel 2a/2b/2c: chunked parallel linear scan  h[s] = dA[s]*h[s-1] + dBu[s]
// Affine-summary composition: (a2,b2)o(a1,b1) = (a2*a1, a2*b1+b2)
// ---------------------------------------------------------------------------
__global__ void s5_scan_chunk(const float* __restrict__ dA,
                              const float* __restrict__ dBu,
                              float* __restrict__ sA,
                              float* __restrict__ sB) {
    const int tid = blockIdx.x * blockDim.x + threadIdx.x;  // (b*CHUNKS+c)*64+n
    if (tid >= BATCH * CHUNKS * DSTATE) return;
    const int n  = tid & (DSTATE - 1);
    const int bc = tid >> 6;
    const int c  = bc & (CHUNKS - 1);
    const int b  = bc / CHUNKS;
    size_t g = ((size_t)b * SEQ + (size_t)c * CLEN) * DSTATE + n;
    float aP = 1.f, bP = 0.f;
    for (int s = 0; s < CLEN; ++s) {
        const float da = dA[g], db = dBu[g];
        bP = da * bP + db;
        aP *= da;
        g += DSTATE;
    }
    sA[tid] = aP;
    sB[tid] = bP;
}

__global__ void s5_scan_combine(const float* __restrict__ sA,
                                const float* __restrict__ sB,
                                float* __restrict__ hin) {
    const int tid = blockIdx.x * blockDim.x + threadIdx.x;  // b*64+n
    if (tid >= BATCH * DSTATE) return;
    const int n = tid & (DSTATE - 1);
    const int b = tid >> 6;
    float h = 0.f;
    for (int c = 0; c < CHUNKS; ++c) {
        const size_t idx = ((size_t)b * CHUNKS + c) * DSTATE + n;
        hin[idx] = h;                        // exclusive carry into chunk c
        h = sA[idx] * h + sB[idx];
    }
}

__global__ void s5_scan_apply(const float* __restrict__ dA,
                              const float* __restrict__ dBu,
                              const float* __restrict__ hin,
                              __bf16* __restrict__ hsb) {
    const int tid = blockIdx.x * blockDim.x + threadIdx.x;
    if (tid >= BATCH * CHUNKS * DSTATE) return;
    const int n  = tid & (DSTATE - 1);
    const int bc = tid >> 6;
    const int c  = bc & (CHUNKS - 1);
    const int b  = bc / CHUNKS;
    size_t g = ((size_t)b * SEQ + (size_t)c * CLEN) * DSTATE + n;
    float h = hin[tid];
    for (int s = 0; s < CLEN; ++s) {
        h = dA[g] * h + dBu[g];
        hsb[g] = (__bf16)h;
        g += DSTATE;
    }
}

// ---------------------------------------------------------------------------
// Kernel 3: one wave per 16-row tile.
//   y[16x512] = hs[16x64](bf16) x C^T(bf16) + D_vec*x   (WMMA, K=64 -> 2 steps)
//   LayerNorm over D=512 with y staged in LDS; coalesced f32 store.
// ---------------------------------------------------------------------------
__global__ __launch_bounds__(32) void s5_gemm2_ln(
        const __bf16* __restrict__ hsb,
        const __bf16* __restrict__ Cb,
        const float* __restrict__ x,
        const float* __restrict__ D_vec,
        const float* __restrict__ gamma,
        const float* __restrict__ beta,
        float* __restrict__ out) {
    __shared__ float ys[16 * DMODEL];
    __shared__ float mus[16];
    __shared__ float rsg[16];

    const int lane = threadIdx.x;
    const int mrow = lane & 15;
    const int half = lane >> 4;
    const int row0 = blockIdx.x * 16;

    // A fragments (hs rows, K = 0..31 and 32..63)
    v16bf a0, a1;
    const __bf16* hrow = hsb + (size_t)(row0 + mrow) * DSTATE;
#pragma unroll
    for (int i = 0; i < 16; ++i) {
        const int c0 = (half << 4) + i;
        a0[i] = hrow[c0];
        a1[i] = hrow[32 + c0];
    }

    float sumr[8], ssqr[8];
#pragma unroll
    for (int r = 0; r < 8; ++r) { sumr[r] = 0.f; ssqr[r] = 0.f; }

    for (int dtile = 0; dtile < DMODEL / 16; ++dtile) {
        const int d = dtile * 16 + mrow;
        // B fragments: b[i] = C_mat[d][kbase + half*16 + i]
        const __bf16* cp = Cb + (size_t)d * DSTATE + (half << 4);
        v16bf b0, b1;
#pragma unroll
        for (int i = 0; i < 16; ++i) { b0[i] = cp[i]; b1[i] = cp[32 + i]; }
        v8f c = {};
        c = __builtin_amdgcn_wmma_f32_16x16x32_bf16(false, a0, false, b0,
                                                    (short)0, c, false, false);
        c = __builtin_amdgcn_wmma_f32_16x16x32_bf16(false, a1, false, b1,
                                                    (short)0, c, false, false);
        const float dv = D_vec[d];
#pragma unroll
        for (int r = 0; r < 8; ++r) {
            const int M = r + (half << 3);
            const float v = c[r] + dv * x[(size_t)(row0 + M) * DMODEL + d];
            ys[M * DMODEL + d] = v;
            sumr[r] += v;
            ssqr[r] += v * v;
        }
    }

    // reduce stats within each 16-lane half (rows 0-7 / 8-15)
#pragma unroll
    for (int r = 0; r < 8; ++r) {
        float s = sumr[r], q = ssqr[r];
#pragma unroll
        for (int o = 1; o < 16; o <<= 1) {
            s += __shfl_xor(s, o);
            q += __shfl_xor(q, o);
        }
        const float mu  = s * (1.f / DMODEL);
        const float var = q * (1.f / DMODEL) - mu * mu;
        if (mrow == 0) {
            mus[r + (half << 3)] = mu;
            rsg[r + (half << 3)] = rsqrtf(var + LN_EPS);
        }
    }
    __syncthreads();

    // normalize + store, coalesced (32 lanes x 16 chunks per row)
    for (int m = 0; m < 16; ++m) {
        const float mu = mus[m];
        const float rs = rsg[m];
        const size_t orow = (size_t)(row0 + m) * DMODEL;
#pragma unroll
        for (int j = 0; j < DMODEL / 32; ++j) {
            const int d = lane + j * 32;
            const float v = ys[m * DMODEL + d];
            out[orow + d] = (v - mu) * rs * gamma[d] + beta[d];
        }
    }
}

// ---------------------------------------------------------------------------
// Host launcher
// inputs: 0:x 1:A_log 2:B_mat 3:C_mat 4:D_vec 5:dt_w 6:dt_b 7:gamma 8:beta
// ---------------------------------------------------------------------------
extern "C" void kernel_launch(void* const* d_in, const int* in_sizes, int n_in,
                              void* d_out, int out_size, void* d_ws, size_t ws_size,
                              hipStream_t stream) {
    (void)in_sizes; (void)n_in; (void)out_size;

    const float* x     = (const float*)d_in[0];
    const float* A_log = (const float*)d_in[1];
    const float* B_mat = (const float*)d_in[2];
    const float* C_mat = (const float*)d_in[3];
    const float* D_vec = (const float*)d_in[4];
    const float* dt_w  = (const float*)d_in[5];
    const float* dt_b  = (const float*)d_in[6];
    const float* gamma = (const float*)d_in[7];
    const float* beta  = (const float*)d_in[8];
    float* out = (float*)d_out;

    // workspace carve-up
    char* ws = (char*)d_ws;
    size_t off = 0;
    float* dA  = (float*)(ws + off);  off += (size_t)ROWS * DSTATE * 4;        // 33.5 MB
    float* dBu = (float*)(ws + off);  off += (size_t)ROWS * DSTATE * 4;        // 33.5 MB
    __bf16* hsb = (__bf16*)(ws + off); off += (size_t)ROWS * DSTATE * 2;       // 16.8 MB
    float* sA  = (float*)(ws + off);  off += (size_t)BATCH * CHUNKS * DSTATE * 4;
    float* sB  = (float*)(ws + off);  off += (size_t)BATCH * CHUNKS * DSTATE * 4;
    float* hin = (float*)(ws + off);  off += (size_t)BATCH * CHUNKS * DSTATE * 4;
    __bf16* Bb = (__bf16*)(ws + off); off += (size_t)DSTATE * DMODEL * 2;
    __bf16* Cb = (__bf16*)(ws + off); off += (size_t)DMODEL * DSTATE * 2;
    if (off > ws_size) return;  // insufficient workspace (deterministic no-op)

    // 0) weight conversion
    s5_cvt_weights<<<(DSTATE * DMODEL + 255) / 256, 256, 0, stream>>>(
        B_mat, C_mat, Bb, Cb);

    // 1) GEMM1 + dt + dA/dBu  (WMMA bf16)
    s5_gemm1_dt<<<RTILES, 32, 0, stream>>>(x, A_log, Bb, dt_w, dt_b, dA, dBu);

    // 2) chunked scan
    const int nscan = BATCH * CHUNKS * DSTATE;   // 131072
    s5_scan_chunk<<<(nscan + 255) / 256, 256, 0, stream>>>(dA, dBu, sA, sB);
    s5_scan_combine<<<(BATCH * DSTATE + 255) / 256, 256, 0, stream>>>(sA, sB, hin);
    s5_scan_apply<<<(nscan + 255) / 256, 256, 0, stream>>>(dA, dBu, hin, hsb);

    // 3) GEMM2 + D*x + LayerNorm (WMMA bf16, LDS-staged LN)
    s5_gemm2_ln<<<RTILES, 32, 0, stream>>>(hsb, Cb, x, D_vec, gamma, beta, out);
}
